// Attention_36739150250680
// MI455X (gfx1250) — matrix operand order repaired and checked by reading, not compiled, not run on hardware
//
#include <hip/hip_runtime.h>

// ---------------- CDNA5 WMMA types ----------------
typedef __attribute__((ext_vector_type(16))) __bf16 v16bf;
typedef __attribute__((ext_vector_type(2)))  __bf16 v2bf;
typedef __attribute__((ext_vector_type(8)))  float  v8f;

#define BATCH 2
#define NSEQ  2048
#define HEADS 16
#define HDIM  64
#define DIM   1024
#define NQKV  3072   // 3*HEADS*HDIM
#define MROWS 4096   // BATCH*NSEQ

__device__ __forceinline__ v8f wmma_bf16(v16bf a, v16bf b, v8f c) {
    return __builtin_amdgcn_wmma_f32_16x16x32_bf16(false, a, false, b, (short)0, c, false, false);
}

// A-operand 16x32 bf16 fragment: lane holds row (lane%16);
// contiguous K = 16*half .. 16*half+15  (two 16B loads)
__device__ __forceinline__ v16bf load_a_frag(const __bf16* row, int k0, int half) {
    v16bf f;
    ((uint4*)&f)[0] = *(const uint4*)(row + k0 + 16 * half);
    ((uint4*)&f)[1] = *(const uint4*)(row + k0 + 16 * half + 8);
    return f;
}

// B-operand 32x16 bf16 fragment from "B-transposed" memory [N][K]:
// lane holds column (lane%16); contiguous K = 16*half .. 16*half+15
__device__ __forceinline__ v16bf load_b_frag(const __bf16* row, int k0, int half) {
    v16bf f;
    const __bf16* p = row + k0 + 16 * half;
    ((uint4*)&f)[0] = *(const uint4*)(p);
    ((uint4*)&f)[1] = *(const uint4*)(p + 8);
    return f;
}

// ---------------- stage 0: precision staging ----------------
__global__ __launch_bounds__(256) void cvt_x_bf16(const float* __restrict__ x,
                                                  __bf16* __restrict__ xb, int n) {
    int i = blockIdx.x * 256 + threadIdx.x;
    if (i < n) xb[i] = (__bf16)x[i];
}

// wT[n][k] = w[k][n]  (contiguous writes; strided reads hit L2)
__global__ __launch_bounds__(256) void cvt_w_t(const float* __restrict__ w,
                                               __bf16* __restrict__ wT) {
    int i = blockIdx.x * 256 + threadIdx.x;   // over NQKV*DIM
    int nn = i >> 10;                          // 0..3071
    int kk = i & 1023;                         // 0..1023
    wT[i] = (__bf16)w[kk * NQKV + nn];
}

// ---------------- stage 1: QKV projection GEMM ----------------
// Each wave: 32x32 output tile, 2x2 accumulators, K-loop step 32.
// Scatters into Q[b,h,n,d], K[b,h,n,d], V^T[b,h,d,n_perm] (bf16).
// V^T columns are permuted within each 32-key block so the attention
// kernel can write packed P pairs:  np = (n&~31)|((n&15)<<1)|((n>>4)&1)
__global__ __launch_bounds__(256) void qkv_gemm(const __bf16* __restrict__ xb,
                                                const __bf16* __restrict__ wT,
                                                __bf16* __restrict__ Qb,
                                                __bf16* __restrict__ Kb,
                                                __bf16* __restrict__ VTb) {
    int wid  = blockIdx.x * 8 + (threadIdx.x >> 5);
    int lane = threadIdx.x & 31;
    int half = lane >> 4, lrow = lane & 15;
    int tm = wid & 127;            // 128 m-tiles
    int tn = wid >> 7;             // 96 n-tiles
    int m0 = tm * 32, n0 = tn * 32;

    const __bf16* a0p = xb + (size_t)(m0 + lrow) * DIM;
    const __bf16* a1p = xb + (size_t)(m0 + 16 + lrow) * DIM;
    const __bf16* b0p = wT + (size_t)(n0 + lrow) * DIM;
    const __bf16* b1p = wT + (size_t)(n0 + 16 + lrow) * DIM;

    v8f c00 = {}, c01 = {}, c10 = {}, c11 = {};
    for (int kk = 0; kk < DIM; kk += 32) {
        v16bf a0 = load_a_frag(a0p, kk, half);
        v16bf a1 = load_a_frag(a1p, kk, half);
        v16bf b0 = load_b_frag(b0p, kk, half);
        v16bf b1 = load_b_frag(b1p, kk, half);
        c00 = wmma_bf16(a0, b0, c00);
        c01 = wmma_bf16(a0, b1, c01);
        c10 = wmma_bf16(a1, b0, c10);
        c11 = wmma_bf16(a1, b1, c11);
    }

#pragma unroll
    for (int mi = 0; mi < 2; ++mi) {
#pragma unroll
        for (int ni = 0; ni < 2; ++ni) {
            v8f c = (mi == 0) ? (ni == 0 ? c00 : c01) : (ni == 0 ? c10 : c11);
            int Ncol  = n0 + ni * 16 + lrow;
            int which = Ncol >> 10;        // 0=Q 1=K 2=V
            int inner = Ncol & 1023;
            int h = inner >> 6, d = inner & 63;
#pragma unroll
            for (int r = 0; r < 8; ++r) {
                int M = m0 + mi * 16 + r + 8 * half;
                int b = M >> 11, n = M & 2047;
                __bf16 val = (__bf16)c[r];
                size_t bh = (size_t)(b * HEADS + h);
                if (which == 0)      Qb [(bh * NSEQ + n) * HDIM + d] = val;
                else if (which == 1) Kb [(bh * NSEQ + n) * HDIM + d] = val;
                else {
                    int np = (n & ~31) | ((n & 15) << 1) | ((n >> 4) & 1);
                    VTb[(bh * HDIM + d) * NSEQ + np] = val;
                }
            }
        }
    }
}

// ---------------- stage 2: flash attention ----------------
// One wave per 16-row q-tile; 64-key strips (4 S-tiles per softmax pass
// to amortize row reductions / O-rescale); online softmax in fp32;
// P relayout through a packed-bf16 LDS tile in permuted key order.
__global__ __launch_bounds__(256) void attn_fwd(const __bf16* __restrict__ Qb,
                                                const __bf16* __restrict__ Kb,
                                                const __bf16* __restrict__ VTb,
                                                float* __restrict__ out) {
    __shared__ __align__(16) __bf16 plds[8][16][64];   // per-wave P strip (C->A relayout)
    int tid  = threadIdx.x;
    int w    = tid >> 5, lane = tid & 31;
    int half = lane >> 4, lrow = lane & 15;
    int gq = blockIdx.x * 8 + w;       // 0..4095
    int qt = gq & 127;                 // q-tile within head
    int bh = gq >> 7;                  // 0..31 (b*HEADS+h)

    const __bf16* Qp = Qb  + ((size_t)bh * NSEQ + qt * 16 + lrow) * HDIM;
    const __bf16* Kp = Kb  + (size_t)bh * NSEQ * HDIM;
    const __bf16* Vp = VTb + (size_t)bh * HDIM * NSEQ;

    v16bf q0 = load_a_frag(Qp, 0,  half);
    v16bf q1 = load_a_frag(Qp, 32, half);

    v8f o0 = {}, o1 = {}, o2 = {}, o3 = {};
    float m[8], l[8];
#pragma unroll
    for (int r = 0; r < 8; ++r) { m[r] = -1e30f; l[r] = 0.f; }
    const float alpha = 0.125f * 1.4426950408889634f;  // 1/sqrt(64) * log2(e)

    for (int kv = 0; kv < NSEQ; kv += 64) {
        // ---- S = Q K^T for 4 key tiles (d=64 -> two k-steps each) ----
        v8f s0 = {}, s1 = {}, s2 = {}, s3 = {};
        const __bf16* kr0 = Kp + (size_t)(kv + lrow) * HDIM;
        const __bf16* kr1 = Kp + (size_t)(kv + 16 + lrow) * HDIM;
        const __bf16* kr2 = Kp + (size_t)(kv + 32 + lrow) * HDIM;
        const __bf16* kr3 = Kp + (size_t)(kv + 48 + lrow) * HDIM;
        s0 = wmma_bf16(q0, load_b_frag(kr0, 0, half), s0);
        s0 = wmma_bf16(q1, load_b_frag(kr0, 32, half), s0);
        s1 = wmma_bf16(q0, load_b_frag(kr1, 0, half), s1);
        s1 = wmma_bf16(q1, load_b_frag(kr1, 32, half), s1);
        s2 = wmma_bf16(q0, load_b_frag(kr2, 0, half), s2);
        s2 = wmma_bf16(q1, load_b_frag(kr2, 32, half), s2);
        s3 = wmma_bf16(q0, load_b_frag(kr3, 0, half), s3);
        s3 = wmma_bf16(q1, load_b_frag(kr3, 32, half), s3);

        // ---- online softmax (row = r + 8*half lives across 16 lanes) ----
#pragma unroll
        for (int r = 0; r < 8; ++r) {
            float v0 = s0[r] * alpha, v1 = s1[r] * alpha;
            float v2 = s2[r] * alpha, v3 = s3[r] * alpha;
            float mr = fmaxf(fmaxf(v0, v1), fmaxf(v2, v3));
            mr = fmaxf(mr, __shfl_xor(mr, 1, 32));
            mr = fmaxf(mr, __shfl_xor(mr, 2, 32));
            mr = fmaxf(mr, __shfl_xor(mr, 4, 32));
            mr = fmaxf(mr, __shfl_xor(mr, 8, 32));
            float mn   = fmaxf(m[r], mr);
            float corr = __builtin_amdgcn_exp2f(m[r] - mn);
            float p0   = __builtin_amdgcn_exp2f(v0 - mn);
            float p1   = __builtin_amdgcn_exp2f(v1 - mn);
            float p2   = __builtin_amdgcn_exp2f(v2 - mn);
            float p3   = __builtin_amdgcn_exp2f(v3 - mn);
            m[r] = mn;
            l[r] = l[r] * corr + ((p0 + p1) + (p2 + p3));
            o0[r] *= corr; o1[r] *= corr; o2[r] *= corr; o3[r] *= corr;
            int M = r + 8 * half;
            // packed pair stores in permuted key order: pos 2*lrow{,+1}
            v2bf pk01; pk01[0] = (__bf16)p0; pk01[1] = (__bf16)p1;
            v2bf pk23; pk23[0] = (__bf16)p2; pk23[1] = (__bf16)p3;
            *(v2bf*)&plds[w][M][2 * lrow]      = pk01;
            *(v2bf*)&plds[w][M][32 + 2 * lrow] = pk23;
        }
        asm volatile("s_wait_dscnt 0" ::: "memory");  // DS in-order per wave; block reordering

        // ---- P (two A-frags over 64 permuted key positions) x V^T ----
        v16bf pf0 = load_a_frag(&plds[w][lrow][0], 0,  half);
        v16bf pf1 = load_a_frag(&plds[w][lrow][0], 32, half);
        const __bf16* vr0 = Vp + (size_t)(lrow)      * NSEQ;
        const __bf16* vr1 = Vp + (size_t)(16 + lrow) * NSEQ;
        const __bf16* vr2 = Vp + (size_t)(32 + lrow) * NSEQ;
        const __bf16* vr3 = Vp + (size_t)(48 + lrow) * NSEQ;
        o0 = wmma_bf16(pf0, load_b_frag(vr0, kv, half), o0);
        o1 = wmma_bf16(pf0, load_b_frag(vr1, kv, half), o1);
        o2 = wmma_bf16(pf0, load_b_frag(vr2, kv, half), o2);
        o3 = wmma_bf16(pf0, load_b_frag(vr3, kv, half), o3);
        o0 = wmma_bf16(pf1, load_b_frag(vr0, kv + 32, half), o0);
        o1 = wmma_bf16(pf1, load_b_frag(vr1, kv + 32, half), o1);
        o2 = wmma_bf16(pf1, load_b_frag(vr2, kv + 32, half), o2);
        o3 = wmma_bf16(pf1, load_b_frag(vr3, kv + 32, half), o3);
        asm volatile("s_wait_dscnt 0" ::: "memory");  // P reads done before next strip's stores
    }

    // ---- finalize: divide by row sums, store [b,h,n,d] fp32 ----
    float inv[8];
#pragma unroll
    for (int r = 0; r < 8; ++r) {
        float ls = l[r];
        ls += __shfl_xor(ls, 1, 32);
        ls += __shfl_xor(ls, 2, 32);
        ls += __shfl_xor(ls, 4, 32);
        ls += __shfl_xor(ls, 8, 32);
        inv[r] = 1.0f / ls;
    }
    float* ob = out + ((size_t)bh * NSEQ + qt * 16) * HDIM;
#pragma unroll
    for (int r = 0; r < 8; ++r) {
        int M = r + 8 * half;
        float* orow = ob + (size_t)M * HDIM;
        orow[lrow]      = o0[r] * inv[r];
        orow[16 + lrow] = o1[r] * inv[r];
        orow[32 + lrow] = o2[r] * inv[r];
        orow[48 + lrow] = o3[r] * inv[r];
    }
}

// ---------------- host launch ----------------
extern "C" void kernel_launch(void* const* d_in, const int* in_sizes, int n_in,
                              void* d_out, int out_size, void* d_ws, size_t ws_size,
                              hipStream_t stream) {
    const float* x = (const float*)d_in[0];   // [2,2048,1024]
    const float* w = (const float*)d_in[1];   // [1024,3072]
    float* out = (float*)d_out;               // [2,16,2048,64]
    char* ws = (char*)d_ws;

    __bf16* xb  = (__bf16*)(ws);                              // 8 MB  [4096][1024]
    __bf16* wT  = (__bf16*)(ws + (size_t)8  * 1024 * 1024);   // 6 MB  [3072][1024]
    __bf16* Qb  = (__bf16*)(ws + (size_t)14 * 1024 * 1024);   // 8 MB  [32][2048][64]
    __bf16* Kb  = (__bf16*)(ws + (size_t)22 * 1024 * 1024);   // 8 MB  [32][2048][64]
    __bf16* VTb = (__bf16*)(ws + (size_t)30 * 1024 * 1024);   // 8 MB  [32][64][2048] (perm cols)

    cvt_x_bf16<<<(MROWS * DIM) / 256, 256, 0, stream>>>(x, xb, MROWS * DIM);
    cvt_w_t  <<<(NQKV * DIM) / 256, 256, 0, stream>>>(w, wT);
    // 128 * 96 tiles of 32x32, 8 waves per block
    qkv_gemm <<<(128 * 96) / 8, 256, 0, stream>>>(xb, wT, Qb, Kb, VTb);
    // 4096 q-tiles, 8 waves per block
    attn_fwd <<<4096 / 8, 256, 0, stream>>>(Qb, Kb, VTb, out);
}
